// LSTM_65472481460517
// MI455X (gfx1250) — matrix operand order repaired and checked by reading, not compiled, not run on hardware
//
#include <hip/hip_runtime.h>
#include <hip/hip_bf16.h>

// ---------------------------------------------------------------------------
// LSTM on MI455X (gfx1250, wave32, WMMA).
//
//  - Persistent 16-WG grid, one global-atomic barrier per timestep.
//  - WG wg owns h-columns [16wg,16wg+16) and all 4 gates for them; wave w
//    owns batch rows [16w,16w+16) -> 4 gate tiles 16x16, K=320 in 10 chunks
//    -> 40 v_wmma_f32_16x16x32_bf16 per wave per step.
//  - A operands (x_t, h_{t-1}) live in global memory already in the bf16
//    WMMA A-fragment layout (pure 32B loads in the hot loop); weights live
//    in LDS in B-fragment layout (40 KB/WG); cell state c stays in VGPRs.
//  - x-projection for step t+1 is computed BEFORE the step-t barrier
//    (independent of h), hiding barrier latency with real GEMM work.
//  - Accumulators start from WMMA's inline C=0; biases are added once in the
//    elementwise tail (and pinned in VGPRs so fences can't force reloads).
//  - Cell nonlinearities use the CDNA5 hardware V_TANH_F32 transcendental;
//    sigmoid(x) = 0.5*tanh(x/2)+0.5.
//  - Non-temporal hints: xfrag loads (stream-once) and h_seq stores
//    (write-once) bypass cache retention; h fragments keep RT (L2 reuse).
//
// Numerics: bf16 operands, fp32 WMMA accumulation (tf32-style GEMM).
// ---------------------------------------------------------------------------

#define T_SEQ 2048
#define BATCH 256
#define DIN   64
#define DH    256
#define NWG   16
#define WGSZ  512   // 16 waves of 32

typedef __attribute__((ext_vector_type(16))) __bf16 v16bf;
typedef __attribute__((ext_vector_type(8)))  float  v8f;

__device__ __forceinline__ float tanh_hw(float x) {
#if __has_builtin(__builtin_amdgcn_tanhf)
    return __builtin_amdgcn_tanhf(x);
#else
    float r;
    // CDNA5 V_TANH_F32; trailing v_nop satisfies the TRANS-op RAW hazard rule
    asm volatile("v_tanh_f32 %0, %1\n\tv_nop" : "=v"(r) : "v"(x));
    return r;
#endif
}
__device__ __forceinline__ float sigf(float x) {
    return __builtin_fmaf(0.5f, tanh_hw(0.5f * x), 0.5f);
}

// ---------------------------------------------------------------------------
// ws layout (bytes):
//   [0,256)            : grid barrier counter (monotonic within a launch)
//   [256, +131072)     : hfrag parity0 : [mtile 16][kchunk 8][lane 32][e 16] bf16
//   [.. , +131072)     : hfrag parity1
//   [.. , +64MB)       : xfrag : [t 2048][mtile 16][kc 2][lane 32][e 16] bf16
// ---------------------------------------------------------------------------

__global__ void lstm_init(unsigned* bar, __bf16* hfrag0) {
    int idx = blockIdx.x * blockDim.x + threadIdx.x;   // 65536 threads
    if (idx == 0) *bar = 0u;
    if (idx < 16 * 8 * 512) hfrag0[idx] = (__bf16)0.0f;   // h0 = 0
}

// Phase 0: x[T,B,IN] fp32 -> bf16 WMMA-A fragments.
// A 16-bit 16x32 layout: lane<16  holds (M=lane,   K: e<8 -> e,   e>=8 -> e+8)
//                        lane>=16 holds (M=lane-16,K: e<8 -> e+8, e>=8 -> e+16)
__global__ void lstm_xprep(const float* __restrict__ x, __bf16* __restrict__ xfrag) {
    const int fi  = blockIdx.x;          // (t*16 + mtile)*2 + kc, 65536 blocks
    const int tid = threadIdx.x;         // 512 = one fragment element each
    const int lane = tid >> 4, e = tid & 15;
    const int kc = fi & 1;
    const int mt = (fi >> 1) & 15;
    const int t  = fi >> 5;
    const int m  = lane & 15;
    int kp;
    if (lane < 16) kp = (e < 8) ? e : e + 8;
    else           kp = (e < 8) ? e + 8 : e + 16;
    const int k = kc * 32 + kp;                         // < 64
    const int b = mt * 16 + m;
    const float v = x[((size_t)t * BATCH + b) * DIN + k];
    xfrag[(size_t)fi * 512 + tid] = (__bf16)v;
}

// x-projection GEMM for one step: acc = x_t @ Wx   (kc 0..1, C starts inline-0)
// Independent of h, so it is called BEFORE the step barrier for t+1.
__device__ __forceinline__ void gemm_xpart(int t, int w, int lane,
                                           const __bf16* __restrict__ xfrag,
                                           const __bf16* lds_w,
                                           v8f acc[4])
{
    v16bf b0[4], b1[4];
#pragma unroll
    for (int gi = 0; gi < 4; gi++)
        b0[gi] = *(const v16bf*)(lds_w + (gi << 9) + lane * 16);
#pragma unroll
    for (int gi = 0; gi < 4; gi++)
        b1[gi] = *(const v16bf*)(lds_w + ((4 + gi) << 9) + lane * 16);

    const v16bf a0 = __builtin_nontemporal_load(
        (const v16bf*)(xfrag + ((size_t)(t * 16 + w) * 2 + 0) * 512 + lane * 16));
    const v16bf a1 = __builtin_nontemporal_load(
        (const v16bf*)(xfrag + ((size_t)(t * 16 + w) * 2 + 1) * 512 + lane * 16));

    const v8f zero = {};   // lowers to WMMA's inline SRC2 = 0
#pragma unroll
    for (int gi = 0; gi < 4; gi++)
        acc[gi] = __builtin_amdgcn_wmma_f32_16x16x32_bf16(
            false, a0, false, b0[gi], (short)0, zero, false, false);
#pragma unroll
    for (int gi = 0; gi < 4; gi++)
        acc[gi] = __builtin_amdgcn_wmma_f32_16x16x32_bf16(
            false, a1, false, b1[gi], (short)0, acc[gi], false, false);
}

__global__ __launch_bounds__(WGSZ, 1)
void lstm_main(const float* __restrict__ Wx, const float* __restrict__ Wh,
               const float* __restrict__ bx, const float* __restrict__ bh,
               const __bf16* __restrict__ xfrag,
               __bf16* __restrict__ hfragA, __bf16* __restrict__ hfragB,
               unsigned* __restrict__ bar,
               float* __restrict__ out)
{
    // Weights in B-fragment layout: [kchunk 10][gate 4][lane 32][e 16] bf16 = 40 KB
    __shared__ __bf16 lds_w[10 * 4 * 512];
    // Per-wave 512B transpose bounce for h writeback
    __shared__ __bf16 lds_sh[16 * 256];

    const int tid  = threadIdx.x;
    const int wg   = blockIdx.x;      // 0..15 : h-column slice
    const int w    = tid >> 5;        // wave id == batch mtile
    const int lane = tid & 31;

    // ---- one-time: stage bf16 weights into LDS (B layout: lane=n(+16 if k'&8), e=(k'&7)+8*(k'>=16))
    for (int idx = tid; idx < 320 * 64; idx += WGSZ) {
        const int k  = idx >> 6, c4 = idx & 63;
        const int gi = c4 >> 4,  c  = c4 & 15;
        const int gcol = gi * 256 + wg * 16 + c;
        const float v = (k < 64) ? Wx[(size_t)k * 1024 + gcol]
                                 : Wh[(size_t)(k - 64) * 1024 + gcol];
        const int kp = k & 31;
        const int lb = c + ((kp & 8) ? 16 : 0);
        lds_w[(((k >> 5) * 4 + gi) << 9) + lb * 16 + ((kp & 7) | ((kp & 16) >> 1))] = (__bf16)v;
    }
    __syncthreads();

    // biases, pinned in VGPRs (opaque asm) so the per-step fences cannot
    // force the compiler to rematerialize the global loads inside the loop
    float bs0, bs1, bs2, bs3;
    {
        const int c = lane & 15;
        bs0 = bx[0 * 256 + wg * 16 + c] + bh[0 * 256 + wg * 16 + c];
        bs1 = bx[1 * 256 + wg * 16 + c] + bh[1 * 256 + wg * 16 + c];
        bs2 = bx[2 * 256 + wg * 16 + c] + bh[2 * 256 + wg * 16 + c];
        bs3 = bx[3 * 256 + wg * 16 + c] + bh[3 * 256 + wg * 16 + c];
        asm volatile("" : "+v"(bs0), "+v"(bs1), "+v"(bs2), "+v"(bs3));
    }

    v8f cst, hv;
#pragma unroll
    for (int r = 0; r < 8; r++) { cst[r] = 0.0f; hv[r] = 0.0f; }

    const int cL   = lane & 15;               // local h column
    const int mb   = (lane >= 16) ? 8 : 0;    // D-layout row base
    const int kcw  = wg >> 1;                 // h kchunk this WG writes
    const int parw = (wg & 1) * 8;            // 16B half within 32B lane chunk

    // x-projection for t=0 (h0 fragments were zeroed by lstm_init)
    v8f acc[4];
    gemm_xpart(0, w, lane, xfrag, lds_w, acc);

    for (int t = 0; t < T_SEQ; ++t) {
        const __bf16* hread  = (t & 1) ? hfragB : hfragA;
        __bf16*       hwrite = (t & 1) ? hfragA : hfragB;

        // ---- recurrent projection: K 64..319 from h_{t-1} fragments,
        //      B fragments software-pipelined from LDS
        {
            v16bf bcur[4], bnxt[4];
#pragma unroll
            for (int gi = 0; gi < 4; gi++)
                bcur[gi] = *(const v16bf*)(lds_w + ((2 * 4 + gi) << 9) + lane * 16);
#pragma unroll
            for (int kc = 2; kc < 10; kc++) {
                const v16bf a = *(const v16bf*)(hread + (((size_t)(w * 8 + (kc - 2))) << 9) + lane * 16);
                if (kc + 1 < 10) {
#pragma unroll
                    for (int gi = 0; gi < 4; gi++)
                        bnxt[gi] = *(const v16bf*)(lds_w + (((kc + 1) * 4 + gi) << 9) + lane * 16);
                }
#pragma unroll
                for (int gi = 0; gi < 4; gi++)
                    acc[gi] = __builtin_amdgcn_wmma_f32_16x16x32_bf16(
                        false, a, false, bcur[gi], (short)0, acc[gi], false, false);
#pragma unroll
                for (int gi = 0; gi < 4; gi++) bcur[gi] = bnxt[gi];
            }
        }

        // ---- LSTM cell (wave-local; hardware v_tanh_f32; bias added here)
#pragma unroll
        for (int r = 0; r < 8; r++) {
            const float f = acc[0][r] + bs0, i = acc[1][r] + bs1;
            const float o = acc[2][r] + bs2, g = acc[3][r] + bs3;
            const float cv = sigf(f) * cst[r] + sigf(i) * tanh_hw(g);
            cst[r] = cv;
            hv[r]  = sigf(o) * tanh_hw(cv);
        }

        // h_seq fp32 output (write-once stream -> non-temporal)
#pragma unroll
        for (int r = 0; r < 8; r++) {
            const int b = w * 16 + mb + r;
            __builtin_nontemporal_store(hv[r],
                out + ((size_t)t * BATCH + b) * DH + wg * 16 + cL);
        }

        // h -> bf16 A-fragment via per-wave LDS transpose bounce.
        // dest A layout for k' = (wg&1)*16 + c:  lane_a = m + 16*(c>=8), byte (c&7)*2
#pragma unroll
        for (int r = 0; r < 8; r++) {
            const int m = mb + r;
            lds_sh[(w << 8) + ((m + ((cL >= 8) ? 16 : 0)) << 3) + (cL & 7)] = (__bf16)hv[r];
        }
        asm volatile("s_wait_dscnt 0" ::: "memory");   // wave-internal DS RAW
        {
            const float4 pkt = *(const float4*)(lds_sh + (w << 8) + lane * 8);
            *(float4*)(hwrite + (((size_t)(w * 8 + kcw)) << 9) + lane * 16 + parw) = pkt;
        }

        // ---- x-projection for t+1 BEFORE the barrier (independent of h):
        //      useful WMMA work while waiting for the slowest WG. acc is dead
        //      after the cell phase, so write straight into it (no copies).
        if (t + 1 < T_SEQ)
            gemm_xpart(t + 1, w, lane, xfrag, lds_w, acc);

        // ---- device-wide step barrier (16 co-resident WGs, monotonic counter)
        __threadfence();
        __syncthreads();
        if (tid == 0) {
            __hip_atomic_fetch_add(bar, 1u, __ATOMIC_RELEASE, __HIP_MEMORY_SCOPE_AGENT);
            const unsigned tgt = (unsigned)(NWG * (t + 1));
            while (__hip_atomic_load(bar, __ATOMIC_ACQUIRE, __HIP_MEMORY_SCOPE_AGENT) < tgt)
                __builtin_amdgcn_s_sleep(2);
        }
        __syncthreads();
        __threadfence();
    }

    // ---- final h, c
    const size_t HSEQ = (size_t)T_SEQ * BATCH * DH;
#pragma unroll
    for (int r = 0; r < 8; r++) {
        const int b = w * 16 + mb + r;
        __builtin_nontemporal_store(hv[r],
            out + HSEQ + (size_t)b * DH + wg * 16 + cL);
        __builtin_nontemporal_store(cst[r],
            out + HSEQ + (size_t)BATCH * DH + (size_t)b * DH + wg * 16 + cL);
    }
}

extern "C" void kernel_launch(void* const* d_in, const int* in_sizes, int n_in,
                              void* d_out, int out_size, void* d_ws, size_t ws_size,
                              hipStream_t stream)
{
    const float* x  = (const float*)d_in[0];
    const float* Wx = (const float*)d_in[1];
    const float* Wh = (const float*)d_in[2];
    const float* bx = (const float*)d_in[3];
    const float* bh = (const float*)d_in[4];
    float* out = (float*)d_out;

    unsigned* bar   = (unsigned*)d_ws;
    __bf16* hfrag0  = (__bf16*)((char*)d_ws + 256);
    __bf16* hfrag1  = hfrag0 + 16 * 8 * 512;
    __bf16* xfrag   = hfrag1 + 16 * 8 * 512;
    // ws needed: 256 + 2*128KB + T*16*2*512*2 B = ~67.4 MB

    (void)in_sizes; (void)n_in; (void)out_size; (void)ws_size;

    lstm_init <<<128, 512, 0, stream>>>(bar, hfrag0);
    lstm_xprep<<<T_SEQ * 16 * 2, 512, 0, stream>>>(x, xfrag);
    lstm_main <<<NWG, WGSZ, 0, stream>>>(Wx, Wh, bx, bh, xfrag, hfrag0, hfrag1, bar, out);
}
